// FavorPlusAttention_81295140979163
// MI455X (gfx1250) — matrix operand order repaired and checked by reading, not compiled, not run on hardware
//
#include <hip/hip_runtime.h>

// ---------------------------------------------------------------------------
// FAVOR+ (Performer) attention block for MI455X / gfx1250, wave32 + WMMA f16,
// with async global->LDS staging (GLOBAL_LOAD_ASYNC_TO_LDS_B128 / ASYNCcnt).
// B=4, N=4096, D=1024, H=16, DH=64, M=266 (padded to Mp=288).
// ---------------------------------------------------------------------------

typedef __attribute__((ext_vector_type(16))) _Float16 v16h;
typedef __attribute__((ext_vector_type(8)))  float    v8f;

#define NSEQ 4096
#define DMODEL 1024
#define NHEAD 16
#define DHEAD 64
#define MFEAT 266
#define MPAD  288
#define NBH   64            // B*H

__device__ __forceinline__ v8f wmma_f16(v16h a, v16h b, v8f c) {
  return __builtin_amdgcn_wmma_f32_16x16x32_f16(
      /*neg_a=*/false, a, /*neg_b=*/false, b,
      /*c_mod=*/(short)0, c, /*reuse_a=*/false, /*reuse_b=*/false);
}

// Async copy 16 bytes global -> LDS without touching VGP' data path.
// LDS byte address = low 32 bits of the generic shared pointer (aperture spec).
__device__ __forceinline__ void async_b128(void* lds_ptr, const void* gptr) {
  unsigned lds_addr = (unsigned)(uintptr_t)lds_ptr;
  asm volatile("global_load_async_to_lds_b128 %0, %1, off"
               :
               : "v"(lds_addr), "v"((unsigned long long)(uintptr_t)gptr)
               : "memory");
}
__device__ __forceinline__ void wait_async0() {
  asm volatile("s_wait_asynccnt 0" ::: "memory");
}

// A fragment (16x32 f16, MxK): lane&15 = row; lanes<16 hold K 0..15,
// lanes>=16 hold K 16..31.  LDS tile is row-major, ldk.
__device__ __forceinline__ v16h load_a_frag(const _Float16* lds, int ldk,
                                            int rowBase, int kOff, int lane) {
  int r  = rowBase + (lane & 15);
  int kb = kOff + ((lane >> 4) << 4);
  const _Float16* p0 = lds + r * ldk + kb;
  const _Float16* p1 = p0 + 8;
  v16h a;
#pragma unroll
  for (int i = 0; i < 8; ++i) { a[i] = p0[i]; a[8 + i] = p1[i]; }
  return a;
}

// B fragment (32x16 f16, KxN): lane&15 = col; lanes<16 hold K 0..15,
// lanes>=16 hold K 16..31.  LDS tile stored transposed: sBT[col][k], ldk.
__device__ __forceinline__ v16h load_b_frag(const _Float16* ldsBT, int ldk,
                                            int colBase, int kOff, int lane) {
  int c  = colBase + (lane & 15);
  int kb = kOff + ((lane >> 4) << 4);
  const _Float16* p = ldsBT + c * ldk + kb;
  v16h b;
#pragma unroll
  for (int i = 0; i < 16; ++i) b[i] = p[i];
  return b;
}

__device__ __forceinline__ unsigned f2ord(float f) {
  unsigned u = __float_as_uint(f);
  return (u & 0x80000000u) ? ~u : (u | 0x80000000u);
}
__device__ __forceinline__ float ord2f(unsigned u) {
  unsigned b = (u & 0x80000000u) ? (u ^ 0x80000000u) : ~u;
  return __uint_as_float(b);
}

// ---------------------------------------------------------------------------
// Generic GEMM: C[row,col] = (sum_k A[row,k] * W[col,k] + bias[col]) * oscale
// out_mode: 0 = f16 row-major, 1 = f32 row-major with ReLU, 2 = f16 to
// head layout [(b*H+h), n, dh] where row=b*N+n, col=h*64+dh.
// Tiles: BM=128, BN=64, BK=32; 8 waves, each owns a 32x32 tile.
// A staging: f16 source -> async global->LDS; f32 source -> convert via VGPRs.
// ---------------------------------------------------------------------------
__global__ __launch_bounds__(256) void gemm_xwt_kernel(
    const void* __restrict__ Aptr, int a_f16,
    const float* __restrict__ W, const float* __restrict__ bias,
    void* __restrict__ Out, int out_mode, float oscale,
    int Rows, int K, int Cols) {
  __shared__ _Float16 sA[128 * 32];
  __shared__ _Float16 sB[64 * 32];   // transposed: sB[col][k]
  const int t = threadIdx.x, lane = t & 31, wave = t >> 5;
  const int wm = (wave & 3) * 32, wn = (wave >> 2) * 32;
  const int m0 = blockIdx.x * 128, n0 = blockIdx.y * 64;
  const float*   Af = (const float*)Aptr;
  const _Float16* Ah = (const _Float16*)Aptr;
  v8f acc[2][2] = {};
  const int ra = t >> 1, kba = (t & 1) * 16;   // A staging: 16 halves / thread
  const int cb2 = t >> 2, kbb = (t & 3) * 8;   // B staging: 8 floats / thread
  for (int k0 = 0; k0 < K; k0 += 32) {
    _Float16* dA = sA + ra * 32 + kba;
    size_t baseA = (size_t)(m0 + ra) * K + k0 + kba;
    if (a_f16) {
      async_b128(dA,     Ah + baseA);
      async_b128(dA + 8, Ah + baseA + 8);
    } else {
      for (int i = 0; i < 16; ++i) dA[i] = (_Float16)Af[baseA + i];
    }
    {
      size_t baseB = (size_t)(n0 + cb2) * K + k0 + kbb;
      _Float16* dB = sB + cb2 * 32 + kbb;
      for (int i = 0; i < 8; ++i) dB[i] = (_Float16)W[baseB + i];
    }
    if (a_f16) wait_async0();
    __syncthreads();
    v16h a0 = load_a_frag(sA, 32, wm, 0, lane);
    v16h a1 = load_a_frag(sA, 32, wm + 16, 0, lane);
    v16h b0 = load_b_frag(sB, 32, wn, 0, lane);
    v16h b1 = load_b_frag(sB, 32, wn + 16, 0, lane);
    acc[0][0] = wmma_f16(a0, b0, acc[0][0]);
    acc[0][1] = wmma_f16(a0, b1, acc[0][1]);
    acc[1][0] = wmma_f16(a1, b0, acc[1][0]);
    acc[1][1] = wmma_f16(a1, b1, acc[1][1]);
    __syncthreads();
  }
#pragma unroll
  for (int i = 0; i < 2; ++i)
#pragma unroll
    for (int j = 0; j < 2; ++j) {
      int col  = n0 + wn + j * 16 + (lane & 15);
      int rowb = m0 + wm + i * 16 + ((lane >> 4) * 8);
      float bv = bias ? bias[col] : 0.0f;
#pragma unroll
      for (int e = 0; e < 8; ++e) {
        int row = rowb + e;
        float v = (acc[i][j][e] + bv) * oscale;
        if (out_mode == 1) {
          ((float*)Out)[(size_t)row * Cols + col] = v > 0.f ? v : 0.f;
        } else if (out_mode == 0) {
          ((_Float16*)Out)[(size_t)row * Cols + col] = (_Float16)v;
        } else {
          int b_ = row >> 12, n = row & (NSEQ - 1);
          int h = col >> 6, dh = col & 63;
          ((_Float16*)Out)[(((size_t)(b_ * NHEAD + h)) * NSEQ + n) * DHEAD + dh] =
              (_Float16)v;
        }
      }
    }
}

// ---------------------------------------------------------------------------
// Feature kernel: td = t_scaled . proj_m (dn pre-folded into qh/kh), then
// FAVOR+ positive features.
// mode 0: reduce global max(td) for keys -> atomicMax(kmaxbits[bh]).
// mode 1: key features  (mx = global key max).
// mode 2: query features (mx = per-row max over valid m).
// Block = (64 rows of n) x (all 288 padded feature cols), per (bh).
// ---------------------------------------------------------------------------
__global__ __launch_bounds__(256) void feat_kernel(
    const _Float16* __restrict__ T, const float* __restrict__ proj,
    _Float16* __restrict__ outP, unsigned int* __restrict__ kmaxbits, int mode) {
  __shared__ _Float16 sT[64 * 64];
  __shared__ _Float16 sP[MPAD * 32];   // transposed: sP[m][k-chunk]
  __shared__ float rowdiag[64];
  __shared__ float rowred[64 * 2];
  __shared__ float rowmx[64];
  __shared__ float wred[8];
  const float RATIO = 0.06131393394849658f;   // 266^-0.5
  const float EPSF  = 1e-4f;
  const int t = threadIdx.x, lane = t & 31, wave = t >> 5;
  const int n0 = blockIdx.x * 64, bh = blockIdx.y;
  const int rb = wave & 3;      // 16-row block within 64 rows
  const int ch = wave >> 2;     // column half (144 cols each)
  {  // async stage T tile (already scaled by dn upstream): 16 halves / thread
    int r = t >> 2, kb = (t & 3) * 16;
    const _Float16* src = T + ((size_t)bh * NSEQ + n0 + r) * DHEAD + kb;
    _Float16* dst = sT + r * 64 + kb;
    async_b128(dst, src);
    async_b128(dst + 8, src + 8);
  }
  wait_async0();
  __syncthreads();
  if (t < 64) {
    float s = 0.f;
    for (int k = 0; k < 64; ++k) { float x = (float)sT[t * 64 + k]; s += x * x; }
    rowdiag[t] = 0.5f * s;      // 0.5*dn^2*sum(t^2) with dn folded into sT
  }
  v8f acc[9] = {};
  for (int kc = 0; kc < 64; kc += 32) {
    __syncthreads();
    for (int idx = t; idx < MPAD * 4; idx += 256) {
      int m = idx >> 2, kb = (idx & 3) * 8;
      _Float16* dst = sP + m * 32 + kb;
      if (m < MFEAT) {
        const float* src = proj + (size_t)m * DHEAD + kc + kb;
        for (int i = 0; i < 8; ++i) dst[i] = (_Float16)src[i];
      } else {
        for (int i = 0; i < 8; ++i) dst[i] = (_Float16)0.f;
      }
    }
    __syncthreads();
    v16h a = load_a_frag(sT, 64, rb * 16, kc, lane);
#pragma unroll
    for (int tj = 0; tj < 9; ++tj) {
      v16h b = load_b_frag(sP, 32, ch * 144 + tj * 16, 0, lane);
      acc[tj] = wmma_f16(a, b, acc[tj]);
    }
  }
  __syncthreads();
  const float NEG = -1e30f;
  float vmax[8];
#pragma unroll
  for (int e = 0; e < 8; ++e) vmax[e] = NEG;
#pragma unroll
  for (int tj = 0; tj < 9; ++tj) {
    int col = ch * 144 + tj * 16 + (lane & 15);
    if (col < MFEAT)
      for (int e = 0; e < 8; ++e) vmax[e] = fmaxf(vmax[e], acc[tj][e]);
  }
  if (mode == 0) {  // global key max
    float m_ = NEG;
    for (int e = 0; e < 8; ++e) m_ = fmaxf(m_, vmax[e]);
    for (int s = 1; s < 32; s <<= 1) m_ = fmaxf(m_, __shfl_xor(m_, s, 32));
    if (lane == 0) wred[wave] = m_;
    __syncthreads();
    if (t == 0) {
      float g = NEG;
      for (int w = 0; w < 8; ++w) g = fmaxf(g, wred[w]);
      atomicMax(kmaxbits + bh, f2ord(g));
    }
    return;
  }
  float mxs = 0.f;
  if (mode == 2) {  // per-row query max: reduce across 16 lanes in each half
    for (int s = 1; s < 16; s <<= 1)
      for (int e = 0; e < 8; ++e)
        vmax[e] = fmaxf(vmax[e], __shfl_xor(vmax[e], s, 32));
    if ((lane & 15) == 0) {
      int rbase = rb * 16 + ((lane >> 4) * 8);
      for (int e = 0; e < 8; ++e) rowred[(rbase + e) * 2 + ch] = vmax[e];
    }
    __syncthreads();
    if (t < 64) rowmx[t] = fmaxf(rowred[t * 2], rowred[t * 2 + 1]);
    __syncthreads();
  } else {
    mxs = ord2f(kmaxbits[bh]);
  }
#pragma unroll
  for (int tj = 0; tj < 9; ++tj) {
    int col = ch * 144 + tj * 16 + (lane & 15);
#pragma unroll
    for (int e = 0; e < 8; ++e) {
      int row = rb * 16 + ((lane >> 4) * 8) + e;
      float outv = 0.f;
      if (col < MFEAT) {
        float mx = (mode == 2) ? rowmx[row] : mxs;
        outv = RATIO * (__expf(acc[tj][e] - rowdiag[row] - mx) + EPSF);
      }
      outP[((size_t)bh * NSEQ + n0 + row) * MPAD + col] = (_Float16)outv;
    }
  }
}

// k_sum[bh][m] = sum_n kp[bh][n][m]
__global__ __launch_bounds__(288) void ksum_kernel(const _Float16* __restrict__ kp,
                                                   float* __restrict__ ksum) {
  int bh = blockIdx.x, m = threadIdx.x;
  const _Float16* p = kp + (size_t)bh * NSEQ * MPAD + m;
  float s = 0.f;
  for (int n = 0; n < NSEQ; ++n) s += (float)p[(size_t)n * MPAD];
  ksum[bh * MPAD + m] = s;
}

// context[bh][m][dh] = sum_n kp[bh][n][m] * v[bh][n][dh]   (288x64, K=4096)
__global__ __launch_bounds__(256) void ctx_kernel(const _Float16* __restrict__ kp,
                                                  const _Float16* __restrict__ vh,
                                                  float* __restrict__ ctx) {
  __shared__ _Float16 sA[MPAD * 32];  // sA[m][kk] = kp[n0+kk][m]  (transposed)
  __shared__ _Float16 sB[64 * 32];    // sB[dh][kk] = v[n0+kk][dh] (transposed)
  const int t = threadIdx.x, lane = t & 31, wave = t >> 5;
  const int bh = blockIdx.x;
  v8f acc[9] = {};
  for (int n0 = 0; n0 < NSEQ; n0 += 32) {
    {
      int kk = t >> 3, mb = (t & 7) * 36;
      const _Float16* src = kp + ((size_t)bh * NSEQ + n0 + kk) * MPAD + mb;
      for (int i = 0; i < 36; ++i) sA[(mb + i) * 32 + kk] = src[i];
    }
    {
      int kk = t >> 3, db = (t & 7) * 8;
      const _Float16* src = vh + ((size_t)bh * NSEQ + n0 + kk) * DHEAD + db;
      for (int i = 0; i < 8; ++i) sB[(db + i) * 32 + kk] = src[i];
    }
    __syncthreads();
#pragma unroll
    for (int it = 0; it < 9; ++it) {
      int ti = wave * 9 + it;
      int cb = ti & 3, rbk = ti >> 2;
      v16h a = load_a_frag(sA, 32, rbk * 16, 0, lane);
      v16h b = load_b_frag(sB, 32, cb * 16, 0, lane);
      acc[it] = wmma_f16(a, b, acc[it]);
    }
    __syncthreads();
  }
#pragma unroll
  for (int it = 0; it < 9; ++it) {
    int ti = wave * 9 + it;
    int cb = ti & 3, rbk = ti >> 2;
    int dh = cb * 16 + (lane & 15);
    int mb = rbk * 16 + ((lane >> 4) * 8);
    for (int e = 0; e < 8; ++e)
      ctx[((size_t)bh * MPAD + mb + e) * DHEAD + dh] = acc[it][e];
  }
}

// d_inv[bh][n] = 1 / sum_m qp[bh][n][m] * ksum[bh][m]
__global__ __launch_bounds__(256) void dinv_kernel(const _Float16* __restrict__ qp,
                                                   const float* __restrict__ ksum,
                                                   float* __restrict__ dinv) {
  __shared__ float sK[MPAD];
  int bh = blockIdx.y;
  int n = blockIdx.x * 256 + threadIdx.x;
  for (int i = threadIdx.x; i < MPAD; i += 256) sK[i] = ksum[bh * MPAD + i];
  __syncthreads();
  const _Float16* row = qp + ((size_t)bh * NSEQ + n) * MPAD;
  float s = 0.f;
  for (int m = 0; m < MPAD; ++m) s += (float)row[m] * sK[m];
  dinv[(size_t)bh * NSEQ + n] = 1.0f / s;
}

// out[bh][n][dh] = (qp[bh][n][:] . ctx[bh][:][dh]) * dinv[bh][n],
// stored merged-head f16: ao[(b*N+n)*D + h*64 + dh]
__global__ __launch_bounds__(256) void outgemm_kernel(
    const _Float16* __restrict__ qp, const float* __restrict__ ctx,
    const float* __restrict__ dinv, _Float16* __restrict__ ao) {
  __shared__ _Float16 sA[128 * 32];
  __shared__ _Float16 sB[64 * 32];    // sB[dh][mk] = ctx[k0+mk][dh]
  const int t = threadIdx.x, lane = t & 31, wave = t >> 5;
  const int wm = (wave & 3) * 32, wn = (wave >> 2) * 32;
  const int n0 = blockIdx.x * 128, bh = blockIdx.y;
  v8f acc[2][2] = {};
  for (int k0 = 0; k0 < MPAD; k0 += 32) {
    {  // async stage qp tile: 16 halves / thread
      int r = t >> 1, kb = (t & 1) * 16;
      const _Float16* src = qp + ((size_t)bh * NSEQ + n0 + r) * MPAD + k0 + kb;
      _Float16* dst = sA + r * 32 + kb;
      async_b128(dst, src);
      async_b128(dst + 8, src + 8);
    }
    {
      int dh = t >> 2, mb = (t & 3) * 8;
      const float* src = ctx + ((size_t)bh * MPAD + k0 + mb) * DHEAD + dh;
      for (int i = 0; i < 8; ++i) sB[dh * 32 + mb + i] = (_Float16)src[i * DHEAD];
    }
    wait_async0();
    __syncthreads();
    v16h a0 = load_a_frag(sA, 32, wm, 0, lane);
    v16h a1 = load_a_frag(sA, 32, wm + 16, 0, lane);
    v16h b0 = load_b_frag(sB, 32, wn, 0, lane);
    v16h b1 = load_b_frag(sB, 32, wn + 16, 0, lane);
    acc[0][0] = wmma_f16(a0, b0, acc[0][0]);
    acc[0][1] = wmma_f16(a0, b1, acc[0][1]);
    acc[1][0] = wmma_f16(a1, b0, acc[1][0]);
    acc[1][1] = wmma_f16(a1, b1, acc[1][1]);
    __syncthreads();
  }
  int b_ = bh >> 4, h = bh & 15;
#pragma unroll
  for (int i = 0; i < 2; ++i)
#pragma unroll
    for (int j = 0; j < 2; ++j) {
      int dh  = wn + j * 16 + (lane & 15);
      int rbb = wm + i * 16 + ((lane >> 4) * 8);
      for (int e = 0; e < 8; ++e) {
        int n = n0 + rbb + e;
        float v = acc[i][j][e] * dinv[(size_t)bh * NSEQ + n];
        ao[((size_t)b_ * NSEQ + n) * DMODEL + h * DHEAD + dh] = (_Float16)v;
      }
    }
}

__global__ void zero_kmax_kernel(unsigned int* p) { p[threadIdx.x] = 0u; }

// ---------------------------------------------------------------------------
// Workspace layout (bytes)
// ---------------------------------------------------------------------------
static constexpr size_t SZ_HEAD = (size_t)NBH * NSEQ * DHEAD * 2;   // 32 MiB f16
static constexpr size_t SZ_P    = (size_t)NBH * NSEQ * MPAD * 2;    // 144 MiB f16
static constexpr size_t OFF_QH   = 0;
static constexpr size_t OFF_KH   = OFF_QH + SZ_HEAD;
static constexpr size_t OFF_VH   = OFF_KH + SZ_HEAD;
static constexpr size_t OFF_QP   = OFF_VH + SZ_HEAD;
static constexpr size_t OFF_KP   = OFF_QP + SZ_P;
static constexpr size_t OFF_KSUM = OFF_KP + SZ_P;
static constexpr size_t OFF_CTX  = OFF_KSUM + (size_t)NBH * MPAD * 4;
static constexpr size_t OFF_DINV = OFF_CTX + (size_t)NBH * MPAD * DHEAD * 4;
static constexpr size_t OFF_AO   = OFF_DINV + (size_t)NBH * NSEQ * 4;
static constexpr size_t OFF_H1   = OFF_AO + SZ_HEAD;
static constexpr size_t OFF_KMAX = OFF_H1 + SZ_HEAD;

extern "C" void kernel_launch(void* const* d_in, const int* in_sizes, int n_in,
                              void* d_out, int out_size, void* d_ws, size_t ws_size,
                              hipStream_t stream) {
  (void)in_sizes; (void)n_in; (void)out_size; (void)ws_size;
  const float* x    = (const float*)d_in[0];
  const float* Wq   = (const float*)d_in[1];
  const float* bq   = (const float*)d_in[2];
  const float* Wk   = (const float*)d_in[3];
  const float* bk   = (const float*)d_in[4];
  const float* Wv   = (const float*)d_in[5];
  const float* bv   = (const float*)d_in[6];
  const float* Wo   = (const float*)d_in[7];
  const float* bo   = (const float*)d_in[8];
  const float* Wfc  = (const float*)d_in[9];
  const float* bfc  = (const float*)d_in[10];
  const float* proj = (const float*)d_in[11];

  char* ws = (char*)d_ws;
  _Float16* qh   = (_Float16*)(ws + OFF_QH);
  _Float16* kh   = (_Float16*)(ws + OFF_KH);
  _Float16* vh   = (_Float16*)(ws + OFF_VH);
  _Float16* qp   = (_Float16*)(ws + OFF_QP);
  _Float16* kp   = (_Float16*)(ws + OFF_KP);
  float*    ksum = (float*)(ws + OFF_KSUM);
  float*    ctx  = (float*)(ws + OFF_CTX);
  float*    dinv = (float*)(ws + OFF_DINV);
  _Float16* ao   = (_Float16*)(ws + OFF_AO);
  _Float16* h1   = (_Float16*)(ws + OFF_H1);
  unsigned int* kmaxbits = (unsigned int*)(ws + OFF_KMAX);

  const int Rows = 4 * NSEQ;  // 16384
  const float DN = 0.35355339059327379f;  // 64^-0.25 folded into q,k

  zero_kmax_kernel<<<1, NBH, 0, stream>>>(kmaxbits);

  dim3 gg(Rows / 128, DMODEL / 64);
  gemm_xwt_kernel<<<gg, 256, 0, stream>>>(x, 0, Wq, bq, qh, 2, DN,  Rows, DMODEL, DMODEL);
  gemm_xwt_kernel<<<gg, 256, 0, stream>>>(x, 0, Wk, bk, kh, 2, DN,  Rows, DMODEL, DMODEL);
  gemm_xwt_kernel<<<gg, 256, 0, stream>>>(x, 0, Wv, bv, vh, 2, 1.f, Rows, DMODEL, DMODEL);

  dim3 gf(NSEQ / 64, NBH);
  feat_kernel<<<gf, 256, 0, stream>>>(kh, proj, nullptr, kmaxbits, 0);  // key max
  feat_kernel<<<gf, 256, 0, stream>>>(kh, proj, kp, kmaxbits, 1);       // kp
  feat_kernel<<<gf, 256, 0, stream>>>(qh, proj, qp, kmaxbits, 2);       // qp

  ksum_kernel<<<NBH, MPAD, 0, stream>>>(kp, ksum);
  ctx_kernel<<<NBH, 256, 0, stream>>>(kp, vh, ctx);
  dinv_kernel<<<dim3(NSEQ / 256, NBH), 256, 0, stream>>>(qp, ksum, dinv);
  outgemm_kernel<<<dim3(NSEQ / 128, NBH), 256, 0, stream>>>(qp, ctx, dinv, ao);

  gemm_xwt_kernel<<<gg, 256, 0, stream>>>(ao, 1, Wo, bo, h1, 0, 1.f, Rows, DMODEL, DMODEL);
  gemm_xwt_kernel<<<gg, 256, 0, stream>>>(h1, 1, Wfc, bfc, d_out, 1, 1.f, Rows, DMODEL, DMODEL);
}
